// MambaLatentBiMap_42863773614416
// MI455X (gfx1250) — matrix operand order repaired and checked by reading, not compiled, not run on hardware
//
#include <hip/hip_runtime.h>
#include <hip/hip_bf16.h>
#include <math.h>

typedef __attribute__((ext_vector_type(16))) _Float16 v16h;
typedef __attribute__((ext_vector_type(4)))  _Float16 h4;
typedef __attribute__((ext_vector_type(8)))  float    v8f;
typedef __attribute__((ext_vector_type(4)))  float    v4f;

#define B_SZ    4
#define D_SZ    256
#define L_SZ    1024
#define DI      256
#define DCONV   4
#define DSTATE  32
#define DTRANK  16

__device__ __forceinline__ float sigmoidf_(float v) { return 1.0f / (1.0f + __expf(-v)); }

// Swap K-groups 8..15 <-> 16..23 within a 32-col A row so each lane's 16-half
// WMMA fragment (K = base..base+7, base+16..base+23) is contiguous in LDS.
__device__ __forceinline__ int apermute(int c) {
    return ((((c >> 3) ^ (c >> 4)) & 1) ? (c ^ 24) : c);
}

// ---------------------------------------------------------------------------
// WMMA GEMM: C[M,N](f32) = A[M,K](f32,row-major,lda) * B (f32)
//   bt==0: B[k,n] at Bp[k*ldb+n];  bt==1: B[k,n] at Bp[n*ldb+k] (transposed)
// Block = 128 threads = 4 waves; block tile 128(M) x 16(N); each wave owns two
// 16x16 tiles sharing one B fragment. K staged in 32-chunks through
// double-buffered LDS as f16 (fragment-contiguous layout -> ds_load_b128).
// Staging is load-all-then-store (MLP) and branchless (clamped addr + select).
// KGUARD: zero-pad K reads (only the K=16 Wdt GEMM; its A has lda>=32 so
// clamped reads stay in-bounds).
// epilogue: 0=none, 1=sigmoid(acc+bias[n]), 2=softplus(acc+bias[n])
// ---------------------------------------------------------------------------
template<bool KGUARD>
__global__ __launch_bounds__(128) void gemm_wmma_f16(
    const float* __restrict__ A, int lda,
    const float* __restrict__ Bp, int ldb, int bt,
    float* __restrict__ C, int ldc,
    int M, int N, int K,
    const float* __restrict__ bias, int epilogue)
{
    __shared__ __align__(32) _Float16 sA[2][128 * 32];   // [buf][row][k] 16 KB
    __shared__ __align__(32) _Float16 sB[2][16 * 32];    // [buf][n][k]          2 KB

    const int t    = threadIdx.x;
    const int lane = t & 31;
    const int wv   = t >> 5;
    const int m0   = blockIdx.y * 128;
    const int n0   = blockIdx.x * 16;

    v8f acc0 = {0.f,0.f,0.f,0.f,0.f,0.f,0.f,0.f};
    v8f acc1 = {0.f,0.f,0.f,0.f,0.f,0.f,0.f,0.f};

    const int mi = lane & 15;            // tile row (A) / col n (B, C)
    const int hi = lane >> 4;            // half-wave select
    const int fragA = hi * 16;           // A fragment offset within row (halves)
    const int kb    = hi * 16;           // B fragment K base

    // loop-invariant per-thread staging coordinates
    const int arow  = t >> 3;            // base A row (0..15), rows arow+16*i
    const int ac0   = (t & 7) * 4;       // A col-in-chunk (4-aligned, group-safe)
    const int aslot = ac0;
    const int bk    = t >> 2;            // non-trans: B row-in-chunk (0..31)
    const int bn4   = (t & 3) * 4;       //            B col group
    const int tn    = t >> 3;            // trans:     B n (0..15)
    const int tk0   = (t & 7) * 4;       //            B col-in-chunk

    auto loadA = [&](int kc, v4f va[8]) {
        int cc = kc + ac0;
        if (KGUARD && cc >= K) cc = 0;                       // clamped, in-bounds
        #pragma unroll
        for (int i = 0; i < 8; ++i)
            va[i] = *reinterpret_cast<const v4f*>(&A[(size_t)(m0 + arow + i * 16) * lda + cc]);
    };
    auto storeA = [&](int buf, int kc, const v4f va[8]) {
        const bool z = KGUARD && (kc + ac0) >= K;
        #pragma unroll
        for (int i = 0; i < 8; ++i) {
            h4 hv;
            #pragma unroll
            for (int j = 0; j < 4; ++j) hv[j] = (_Float16)(z ? 0.0f : va[i][j]);
            *reinterpret_cast<h4*>(&sA[buf][(arow + i * 16) * 32 + aslot]) = hv;
        }
    };
    auto loadB = [&](int kc, v4f& vb) {
        if (!bt) {
            int kk = kc + bk;
            if (KGUARD && kk >= K) kk = 0;
            vb = *reinterpret_cast<const v4f*>(&Bp[(size_t)kk * ldb + n0 + bn4]);
        } else {
            int cc = kc + tk0;
            if (KGUARD && cc >= K) cc = 0;
            vb = *reinterpret_cast<const v4f*>(&Bp[(size_t)(n0 + tn) * ldb + cc]);
        }
    };
    auto storeB = [&](int buf, int kc, v4f vb) {
        if (!bt) {
            const bool z = KGUARD && (kc + bk) >= K;
            #pragma unroll
            for (int j = 0; j < 4; ++j)
                sB[buf][(bn4 + j) * 32 + bk] = (_Float16)(z ? 0.0f : vb[j]);
        } else {
            const bool z = KGUARD && (kc + tk0) >= K;
            h4 hv;
            #pragma unroll
            for (int j = 0; j < 4; ++j) hv[j] = (_Float16)(z ? 0.0f : vb[j]);
            *reinterpret_cast<h4*>(&sB[buf][tn * 32 + tk0]) = hv;
        }
    };

    // prologue: stage chunk 0
    {
        v4f va[8]; v4f vb;
        loadA(0, va); loadB(0, vb);
        storeA(0, 0, va); storeB(0, 0, vb);
    }
    __syncthreads();

    for (int kc = 0; kc < K; kc += 32) {
        const int  cur  = (kc >> 5) & 1;
        const bool more = (kc + 32) < K;                     // wave-uniform

        v4f va[8]; v4f vb;
        if (more) { loadA(kc + 32, va); loadB(kc + 32, vb); }
        if (kc + 64 < K)                                     // prefetch chunk after next
            __builtin_prefetch(&A[(size_t)(m0 + t) * lda + kc + 64], 0, 1);

        v16h af0 = *reinterpret_cast<const v16h*>(&sA[cur][(wv * 32 +      mi) * 32 + fragA]);
        v16h af1 = *reinterpret_cast<const v16h*>(&sA[cur][(wv * 32 + 16 + mi) * 32 + fragA]);
        v16h bf  = *reinterpret_cast<const v16h*>(&sB[cur][mi * 32 + kb]);

        acc0 = __builtin_amdgcn_wmma_f32_16x16x32_f16(false, af0, false, bf,
                                                      (short)0, acc0, false, false);
        acc1 = __builtin_amdgcn_wmma_f32_16x16x32_f16(false, af1, false, bf,
                                                      (short)0, acc1, false, false);

        if (more) { storeA(cur ^ 1, kc + 32, va); storeB(cur ^ 1, kc + 32, vb); }
        __syncthreads();
    }

    // C/D layout: VGPR v -> row v + 8*hi, col = lane&15
    const int   col  = n0 + mi;
    const int   rb0  = m0 + wv * 32 + hi * 8;
    const float bval = bias ? bias[col] : 0.0f;
    #pragma unroll
    for (int v = 0; v < 8; ++v) {
        float o0 = acc0[v], o1 = acc1[v];
        if (epilogue == 1) {
            o0 = sigmoidf_(o0 + bval);
            o1 = sigmoidf_(o1 + bval);
        } else if (epilogue == 2) {
            float s0 = o0 + bval; o0 = (s0 > 20.f) ? s0 : log1pf(__expf(s0));
            float s1 = o1 + bval; o1 = (s1 > 20.f) ? s1 : log1pf(__expf(s1));
        }
        C[(size_t)(rb0 + v)      * ldc + col] = o0;
        C[(size_t)(rb0 + 16 + v) * ldc + col] = o1;
    }
}

// ---------------------------------------------------------------------------
// (B,D,L) -> (B,L,D), optional flip along L
// ---------------------------------------------------------------------------
__global__ void transpose_flip(const float* __restrict__ x, float* __restrict__ out, int flip)
{
    int idx = blockIdx.x * blockDim.x + threadIdx.x;       // over B*L*D
    int d = idx & (D_SZ - 1);
    int l = (idx >> 8) & (L_SZ - 1);
    int b = idx >> 18;
    int ls = flip ? (L_SZ - 1 - l) : l;
    out[idx] = x[((size_t)b * D_SZ + d) * L_SZ + ls];
}

// ---------------------------------------------------------------------------
// Causal depthwise conv (width 4) + bias + SiLU. uz is (B,L,2*DI); uses [0,DI).
// ---------------------------------------------------------------------------
__global__ void conv_silu(const float* __restrict__ uz, const float* __restrict__ Wc,
                          const float* __restrict__ bc, float* __restrict__ u)
{
    int idx = blockIdx.x * blockDim.x + threadIdx.x;       // over B*L*DI
    int d = idx & (DI - 1);
    int l = (idx >> 8) & (L_SZ - 1);
    int b = idx >> 18;
    float acc = bc[d];
    #pragma unroll
    for (int k = 0; k < DCONV; ++k) {
        int lp = l - (DCONV - 1) + k;
        if (lp >= 0)
            acc += uz[((size_t)(b * L_SZ + lp)) * (2 * DI) + d] * Wc[d * DCONV + k];
    }
    u[idx] = acc * sigmoidf_(acc);
}

// ---------------------------------------------------------------------------
// Selective scan: one wave32 per (b,d) channel, lane = state index n.
// h_n <- exp(dt*A_n)*h_n + dt*u*B_n ;  y = sum_n h_n*C_n + u*Dskip ; y *= silu(z)
// ---------------------------------------------------------------------------
__global__ __launch_bounds__(256) void selective_scan(
    const float* __restrict__ ubuf, const float* __restrict__ dtb,
    const float* __restrict__ proj, const float* __restrict__ uz,
    const float* __restrict__ Alog, const float* __restrict__ Dskip,
    float* __restrict__ yb)
{
    int wid  = blockIdx.x * 8 + (threadIdx.x >> 5);        // 0 .. B*DI-1
    int lane = threadIdx.x & 31;                           // state n
    int b = wid >> 8;
    int d = wid & (DI - 1);

    float An = -__expf(Alog[d * DSTATE + lane]);
    float Dd = Dskip[d];
    float h  = 0.0f;

    for (int l = 0; l < L_SZ; ++l) {
        size_t row = (size_t)(b * L_SZ + l);
        float dt = dtb[row * DI + d];                      // wave-uniform load
        float uv = ubuf[row * DI + d];
        float Bn = proj[row * 80 + DTRANK + lane];
        float Cn = proj[row * 80 + DTRANK + DSTATE + lane];

        h = expf(dt * An) * h + dt * uv * Bn;
        float contrib = h * Cn;
        #pragma unroll
        for (int off = 16; off > 0; off >>= 1)
            contrib += __shfl_xor(contrib, off, 32);

        if (lane == 0) {
            float z = uz[row * (2 * DI) + DI + d];
            float y = contrib + uv * Dd;
            yb[row * DI + d] = y * (z * sigmoidf_(z));
        }
    }
}

// ---------------------------------------------------------------------------
// branch output:  gamma[l]*tanh(alpha*h) + beta[l]   over (B,L,D) buffer
// ---------------------------------------------------------------------------
__global__ void branch_epilogue(const float* __restrict__ lout,
                                const float* __restrict__ gamma,
                                const float* __restrict__ beta,
                                const float* __restrict__ alpha,
                                float* __restrict__ br)
{
    int idx = blockIdx.x * blockDim.x + threadIdx.x;       // over B*L*D
    int l = (idx >> 8) & (L_SZ - 1);
    float a = alpha[0];
    br[idx] = gamma[l] * tanhf(a * lout[idx]) + beta[l];
}

// ---------------------------------------------------------------------------
// out[b,d,l] = br0[b,l,d]*g0[b,d,l] + br1[b,L-1-l,d]*g1[b,d,L-1-l]
// ---------------------------------------------------------------------------
__global__ void combine(const float* __restrict__ br0, const float* __restrict__ g0,
                        const float* __restrict__ br1, const float* __restrict__ g1,
                        float* __restrict__ out)
{
    int idx = blockIdx.x * blockDim.x + threadIdx.x;       // over B*D*L (output layout)
    int l = idx & (L_SZ - 1);
    int d = (idx >> 10) & (D_SZ - 1);
    int b = idx >> 18;
    int lr = L_SZ - 1 - l;
    float m  = br0[((size_t)(b * L_SZ + l))  * D_SZ + d] * g0[((size_t)(b * D_SZ + d)) * L_SZ + l];
    float rm = br1[((size_t)(b * L_SZ + lr)) * D_SZ + d] * g1[((size_t)(b * D_SZ + d)) * L_SZ + lr];
    out[idx] = m + rm;
}

// ---------------------------------------------------------------------------
extern "C" void kernel_launch(void* const* d_in, const int* in_sizes, int n_in,
                              void* d_out, int out_size, void* d_ws, size_t ws_size,
                              hipStream_t stream)
{
    (void)in_sizes; (void)n_in; (void)out_size; (void)ws_size;

    const float* x     = (const float*)d_in[0];
    const float* Win   = (const float*)d_in[1];
    const float* Wconv = (const float*)d_in[2];
    const float* bconv = (const float*)d_in[3];
    const float* Wxp   = (const float*)d_in[4];
    const float* Wdt   = (const float*)d_in[5];
    const float* bdt   = (const float*)d_in[6];
    const float* Alog  = (const float*)d_in[7];
    const float* Dskip = (const float*)d_in[8];
    const float* Wout  = (const float*)d_in[9];
    const float* Wg    = (const float*)d_in[10];
    const float* bg    = (const float*)d_in[11];
    const float* alpha = (const float*)d_in[12];
    const float* gamma = (const float*)d_in[13];
    const float* beta  = (const float*)d_in[14];

    float* ws = (float*)d_ws;
    const size_t BLD = (size_t)B_SZ * L_SZ * D_SZ;         // 1 Mi elements

    float* xln  = ws;                                      // (B,L,D)
    float* uz   = xln  + BLD;                              // (B,L,2*DI)
    float* ub   = uz   + 2 * BLD;                          // (B,L,DI)
    float* proj = ub   + BLD;                              // (B,L,80)
    float* dtb  = proj + (size_t)B_SZ * L_SZ * 80;         // (B,L,DI)
    float* yb   = dtb  + BLD;                              // (B,L,DI)
    float* lout = yb   + BLD;                              // (B,L,D)
    float* br0  = lout + BLD;
    float* br1  = br0  + BLD;
    float* g0   = br1  + BLD;                              // (B*D, L)
    float* g1   = g0   + BLD;                              // total ~47.4 MB

    const int ML = B_SZ * L_SZ;                            // 4096 GEMM rows

    auto gemm = [&](const float* A, int lda, const float* Bp, int ldb, int bt,
                    float* Cp, int ldc, int M, int N, int K,
                    const float* bias, int ep, bool kguard) {
        dim3 grid(N / 16, M / 128);
        if (kguard)
            gemm_wmma_f16<true ><<<grid, dim3(128), 0, stream>>>(A, lda, Bp, ldb, bt,
                                                                 Cp, ldc, M, N, K, bias, ep);
        else
            gemm_wmma_f16<false><<<grid, dim3(128), 0, stream>>>(A, lda, Bp, ldb, bt,
                                                                 Cp, ldc, M, N, K, bias, ep);
    };

    auto run_layer = [&](const float* inp, float* outp, int i) {
        // uz = inp @ Win[i]
        gemm(inp, D_SZ, Win + (size_t)i * D_SZ * 2 * DI, 2 * DI, 0,
             uz, 2 * DI, ML, 2 * DI, D_SZ, nullptr, 0, false);
        // depthwise conv + silu
        conv_silu<<<(int)(BLD / 256), 256, 0, stream>>>(
            uz, Wconv + (size_t)i * DI * DCONV, bconv + (size_t)i * DI, ub);
        // proj = u @ Wxp[i]  -> [dtp | Bm | Cm]
        gemm(ub, DI, Wxp + (size_t)i * DI * 80, 80, 0,
             proj, 80, ML, 80, DI, nullptr, 0, false);
        // dt = softplus(dtp @ Wdt[i] + bdt[i])   (K=16 -> clamped/zero-padded)
        gemm(proj, 80, Wdt + (size_t)i * DTRANK * DI, DI, 0,
             dtb, DI, ML, DI, DTRANK, bdt + (size_t)i * DI, 2, true);
        // selective scan + Dskip + silu(z) gating
        selective_scan<<<128, 256, 0, stream>>>(
            ub, dtb, proj, uz,
            Alog + (size_t)i * DI * DSTATE, Dskip + (size_t)i * DI, yb);
        // out = y @ Wout[i]
        gemm(yb, DI, Wout + (size_t)i * DI * D_SZ, D_SZ, 0,
             outp, D_SZ, ML, D_SZ, DI, nullptr, 0, false);
    };

    float* br[2] = { br0, br1 };
    float* gt[2] = { g0, g1 };

    for (int j = 0; j < 2; ++j) {
        transpose_flip<<<(int)(BLD / 256), 256, 0, stream>>>(x, xln, j);
        run_layer(xln,  lout, 2 * j);
        run_layer(lout, lout, 2 * j + 1);
        branch_epilogue<<<(int)(BLD / 256), 256, 0, stream>>>(
            lout, gamma + (size_t)j * L_SZ, beta + (size_t)j * L_SZ, alpha + j, br[j]);
        // gate[j] = sigmoid(x_rows @ Wg[j]^T + bg[j]) : A=(B*D,L), B transposed
        gemm(x, L_SZ, Wg + (size_t)j * L_SZ * L_SZ, L_SZ, 1,
             gt[j], L_SZ, B_SZ * D_SZ, L_SZ, L_SZ, bg + (size_t)j * L_SZ, 1, false);
    }

    combine<<<(int)(BLD / 256), 256, 0, stream>>>(br[0], gt[0], br[1], gt[1], (float*)d_out);
}
